// ModulatedConv1d_37598143709187
// MI455X (gfx1250) — matrix operand order repaired and checked by reading, not compiled, not run on hardware
//
#include <hip/hip_runtime.h>
#include <hip/hip_bf16.h>

typedef _Float16 half16 __attribute__((ext_vector_type(16)));
typedef _Float16 half8  __attribute__((ext_vector_type(8)));
typedef float    f32x8  __attribute__((ext_vector_type(8)));

#define B_   16
#define C_   512
#define T_   4096
#define S_   512
#define K_   3
#define KK_  (C_ * K_)     // 1536: GEMM K, ordered kk = k*512 + cin

// ---------------------------------------------------------------------------
// CDNA5 async global->LDS copy (16 B per lane), tracked by ASYNCcnt.
// lds_off is the wave-relative LDS byte offset (= low 32 bits of flat addr).
// ---------------------------------------------------------------------------
__device__ __forceinline__ void async_copy_b128(unsigned lds_off, const void* g) {
    asm volatile("global_load_async_to_lds_b128 %0, %1, off"
                 :: "v"(lds_off), "v"((unsigned long long)(uintptr_t)g)
                 : "memory");
}
__device__ __forceinline__ void wait_async0() {
    asm volatile("s_wait_asynccnt 0" ::: "memory");
}

// ---------------------------------------------------------------------------
// Kernel 1: style projection  s[b][c] = styles[b]·proj_w[c] + proj_b[c]
// ---------------------------------------------------------------------------
__global__ __launch_bounds__(256) void style_proj_kernel(
    const float* __restrict__ styles, const float* __restrict__ proj_w,
    const float* __restrict__ proj_b, float* __restrict__ s_out)
{
    int idx = blockIdx.x * 256 + threadIdx.x;       // [0, B*C)
    int b = idx >> 9;
    int c = idx & (C_ - 1);
    const float* st = styles + b * S_;
    const float* pw = proj_w + (size_t)c * S_;
    float acc = 0.f;
    #pragma unroll 8
    for (int j = 0; j < S_; ++j) acc += st[j] * pw[j];
    s_out[idx] = acc + proj_b[c];
}

// ---------------------------------------------------------------------------
// Kernel 2: modulate + demodulate + cast to f16 in GEMM layout
//   wmod[b][cout][k*512 + cin] = f16( W[cout][cin][k] * s[b][cin] * rsqrt(ss) )
// ---------------------------------------------------------------------------
__global__ __launch_bounds__(256) void modulate_kernel(
    const float* __restrict__ weights, const float* __restrict__ s_in,
    _Float16* __restrict__ wmod)
{
    const int bc   = blockIdx.x;            // b*512 + cout
    const int b    = bc >> 9;
    const int cout = bc & (C_ - 1);
    const float* wrow = weights + (size_t)cout * (C_ * K_);   // [cin][k]
    const float* srow = s_in + b * C_;

    float v[6];
    float sq = 0.f;
    #pragma unroll
    for (int i = 0; i < 6; ++i) {
        int kk  = threadIdx.x + i * 256;    // kk = k*512 + cin
        int k   = kk >> 9;
        int cin = kk & (C_ - 1);
        float val = wrow[cin * K_ + k] * srow[cin];
        v[i] = val;
        sq += val * val;
    }

    __shared__ float red[256];
    red[threadIdx.x] = sq;
    __syncthreads();
    #pragma unroll
    for (int off = 128; off > 0; off >>= 1) {
        if ((int)threadIdx.x < off) red[threadIdx.x] += red[threadIdx.x + off];
        __syncthreads();
    }
    float scale = rsqrtf(red[0] + 1e-8f);

    _Float16* orow = wmod + (size_t)bc * KK_;
    #pragma unroll
    for (int i = 0; i < 6; ++i) {
        int kk = threadIdx.x + i * 256;
        orow[kk] = (_Float16)(v[i] * scale);
    }
}

// ---------------------------------------------------------------------------
// Kernel 2b: transpose + convert  x[b][c][t] f32  ->  xT[b][t][c] f16
// 32x32 LDS tiles; coalesced on both sides.
// ---------------------------------------------------------------------------
__global__ __launch_bounds__(256) void transpose_x_kernel(
    const float* __restrict__ x, _Float16* __restrict__ xT)
{
    __shared__ float tile[32][33];
    const int b  = blockIdx.z;
    const int c0 = blockIdx.y * 32;
    const int t0 = blockIdx.x * 32;
    const int col = threadIdx.x & 31;
    const int row0 = threadIdx.x >> 5;          // 0..7

    const float* src = x + ((size_t)(b * C_ + c0)) * T_ + t0;
    #pragma unroll
    for (int i = 0; i < 4; ++i) {
        int r = row0 + i * 8;                   // channel within tile
        tile[r][col] = src[(size_t)r * T_ + col];
    }
    __syncthreads();
    _Float16* dst = xT + ((size_t)b * T_ + t0) * C_ + c0;
    #pragma unroll
    for (int i = 0; i < 4; ++i) {
        int r = row0 + i * 8;                   // time within tile
        dst[(size_t)r * C_ + col] = (_Float16)tile[col][r];
    }
}

// ---------------------------------------------------------------------------
// Kernel 3: batched GEMM-as-conv with WMMA f16 -> f32, double buffered,
// fully async-staged (both A and X tiles via GLOBAL_LOAD_ASYNC_TO_LDS_B128).
//   Tile: 256 (M=cout) x 128 (N=t) per 256-thread WG, K-step 32, 48 steps.
//   8 waves in a 4x2 grid; each wave owns 64x64 = 4x4 WMMA accumulators.
// ---------------------------------------------------------------------------
#define TM     256
#define TN     128
#define KB     32
#define LDA    40              // padded LDS row stride in halfs (80 B)
#define NSTEPS (KK_ / KB)      // 48

__global__ __launch_bounds__(256) void conv_wmma_kernel(
    const _Float16* __restrict__ xT, const _Float16* __restrict__ wmod,
    float* __restrict__ out)
{
    __shared__ _Float16 As[2][TM * LDA];   // As[buf][coutLocal][kkLocal]
    __shared__ _Float16 Xs[2][TN * LDA];   // Xs[buf][n][kkLocal]

    const int b     = blockIdx.z;
    const int cout0 = blockIdx.y * TM;
    const int t0    = blockIdx.x * TN;

    const int tid   = threadIdx.x;
    const int lane  = tid & 31;
    const int wave  = tid >> 5;
    const int waveM = wave >> 1;        // 0..3  (64 rows each)
    const int waveN = wave & 1;         // 0..1  (64 cols each)
    const int m15   = lane & 15;
    const int hi    = lane >> 4;

    const _Float16* Ag  = wmod + ((size_t)(b * C_ + cout0)) * KK_;
    const _Float16* XTg = xT + (size_t)b * T_ * C_;

    f32x8 acc[4][4];
    #pragma unroll
    for (int mi = 0; mi < 4; ++mi)
        #pragma unroll
        for (int ni = 0; ni < 4; ++ni)
            acc[mi][ni] = (f32x8)0.f;

    // staging coordinates
    const int aRow = tid;                 // 256 A rows, one per thread (64 B)
    const int xN   = tid & (TN - 1);      // 128 X cols, 2 threads per col
    const int xSel = (tid >> 7) * 16;     // 16 cin each (32 B)

    const unsigned asLds[2] = {
        (unsigned)(uintptr_t)&As[0][aRow * LDA],
        (unsigned)(uintptr_t)&As[1][aRow * LDA] };
    const unsigned xsLds[2] = {
        (unsigned)(uintptr_t)&Xs[0][xN * LDA + xSel],
        (unsigned)(uintptr_t)&Xs[1][xN * LDA + xSel] };

    // ---- issue async A-tile copy for step s into buffer bf ----
    auto issueA = [&](int s, int bf) {
        const _Float16* g = Ag + (size_t)aRow * KK_ + s * KB;
        #pragma unroll
        for (int c = 0; c < 4; ++c)
            async_copy_b128(asLds[bf] + c * 16, g + c * 8);
    };
    // ---- issue async X-tile copy (32 B contiguous in xT) ----
    auto issueX = [&](int s, int bf) {
        const int ks   = s * KB;
        const int k    = ks >> 9;
        const int cin0 = ks & (C_ - 1);
        const int tg   = t0 + xN + k - 1;          // conv 'same' padding
        if ((unsigned)tg < (unsigned)T_) {
            const _Float16* g = XTg + (size_t)tg * C_ + cin0 + xSel;
            async_copy_b128(xsLds[bf],      g);
            async_copy_b128(xsLds[bf] + 16, g + 8);
        } else {
            half16 z = {};
            *(half16*)&Xs[bf][xN * LDA + xSel] = z;   // zero-pad row
        }
    };
    // ---- fragment loads (ISA 7.12.2 layouts) + 16 WMMAs on buffer p ----
    auto compute = [&](int p) {
        half16 af[4];
        #pragma unroll
        for (int mi = 0; mi < 4; ++mi) {
            const _Float16* q = &As[p][(waveM * 64 + mi * 16 + m15) * LDA];
            half8 c0 = *(const half8*)(q + hi * 8);        // K 0-7  / 8-15
            half8 c1 = *(const half8*)(q + 16 + hi * 8);   // K 16-23 / 24-31
            af[mi] = __builtin_shufflevector(c0, c1,
                        0,1,2,3,4,5,6,7,8,9,10,11,12,13,14,15);
        }
        half16 bv[4];
        #pragma unroll
        for (int ni = 0; ni < 4; ++ni) {
            const _Float16* q = &Xs[p][(waveN * 64 + ni * 16 + m15) * LDA];
            half8 d0 = *(const half8*)(q + hi * 8);        // K 0-7  / 8-15
            half8 d1 = *(const half8*)(q + 16 + hi * 8);   // K 16-23 / 24-31
            bv[ni] = __builtin_shufflevector(d0, d1,
                        0,1,2,3,4,5,6,7,8,9,10,11,12,13,14,15);
        }
        #pragma unroll
        for (int mi = 0; mi < 4; ++mi)
            #pragma unroll
            for (int ni = 0; ni < 4; ++ni)
                acc[mi][ni] = __builtin_amdgcn_wmma_f32_16x16x32_f16(
                    false, af[mi], false, bv[ni],
                    (short)0, acc[mi][ni], false, false);
    };

    // ---- prologue: fill buffer 0 ----
    issueA(0, 0);
    issueX(0, 0);
    wait_async0();
    __syncthreads();

    // ---- main pipelined loop ----
    for (int s = 0; s < NSTEPS; ++s) {
        const int p = s & 1;
        if (s + 1 < NSTEPS) {            // prefetch next tile (async)
            issueA(s + 1, p ^ 1);
            issueX(s + 1, p ^ 1);
        }
        compute(p);
        wait_async0();                   // async writes for p^1 landed
        __syncthreads();                 // publish p^1, retire reads of p
    }

    // ---- epilogue: C/D layout  M = hi*8 + r, N = lane&15 ----
    float* Og = out + ((size_t)(b * C_ + cout0)) * T_ + t0;
    #pragma unroll
    for (int mi = 0; mi < 4; ++mi) {
        #pragma unroll
        for (int ni = 0; ni < 4; ++ni) {
            const int mb = waveM * 64 + mi * 16 + hi * 8;
            const int nb = waveN * 64 + ni * 16 + m15;
            #pragma unroll
            for (int r = 0; r < 8; ++r)
                Og[(size_t)(mb + r) * T_ + nb] = acc[mi][ni][r];
        }
    }
}

// ---------------------------------------------------------------------------
extern "C" void kernel_launch(void* const* d_in, const int* in_sizes, int n_in,
                              void* d_out, int out_size, void* d_ws, size_t ws_size,
                              hipStream_t stream) {
    const float* inputs  = (const float*)d_in[0];   // [B, C, T]
    const float* styles  = (const float*)d_in[1];   // [B, S]
    const float* proj_w  = (const float*)d_in[2];   // [C, S]
    const float* proj_b  = (const float*)d_in[3];   // [C]
    const float* weights = (const float*)d_in[4];   // [1, C, C, K]
    float* out = (float*)d_out;                     // [B, C, T]

    // workspace: s (32 KB) | wmod (B*C*KK f16, ~25.2 MB) | xT (B*T*C f16, 64 MB)
    float*     s_ws = (float*)d_ws;
    _Float16*  wmod = (_Float16*)((char*)d_ws + (size_t)B_ * C_ * sizeof(float));
    _Float16*  xTws = (_Float16*)((char*)wmod + (size_t)B_ * C_ * KK_ * sizeof(_Float16));

    style_proj_kernel<<<(B_ * C_) / 256, 256, 0, stream>>>(
        styles, proj_w, proj_b, s_ws);

    modulate_kernel<<<B_ * C_, 256, 0, stream>>>(weights, s_ws, wmod);

    dim3 tgrid(T_ / 32, C_ / 32, B_);
    transpose_x_kernel<<<tgrid, 256, 0, stream>>>(inputs, xTws);

    dim3 grid(T_ / TN, C_ / TM, B_);
    conv_wmma_kernel<<<grid, 256, 0, stream>>>(xTws, wmod, out);
}
